// MultiHeadGraphAttention_weight_79431125172539
// MI455X (gfx1250) — matrix operand order repaired and checked by reading, not compiled, not run on hardware
//
#include <hip/hip_runtime.h>

#define NN     4096
#define FIN    1024
#define FOUT   128
#define NHEAD  8
#define LEAKY  0.2f
#define NEGINF (-9e15f)

typedef __attribute__((ext_vector_type(16))) __bf16 v16bf;
typedef __attribute__((ext_vector_type(8)))  float  v8f;

union FragU { v16bf v; unsigned short us[16]; uint4 q[2]; };

__device__ inline unsigned short f2bf(float x) {
  unsigned int u = __float_as_uint(x);
  u += 0x7FFFu + ((u >> 16) & 1u);          // round-to-nearest-even
  return (unsigned short)(u >> 16);
}
__device__ inline float bf2f(unsigned short b) {
  return __uint_as_float(((unsigned int)b) << 16);
}

// ---------------- prep: fp32 -> bf16 conversions ----------------
__global__ void gat_prep_h(const float* __restrict__ h, unsigned short* __restrict__ hb) {
  int idx = blockIdx.x * 256 + threadIdx.x;          // NN*FIN elements
  hb[idx] = f2bf(h[idx]);
}

__global__ void gat_prep_w(const float* __restrict__ w, unsigned short* __restrict__ wT) {
  int idx = blockIdx.x * 256 + threadIdx.x;          // NHEAD*FIN*FOUT elements
  int o  = idx % FOUT;
  int k  = (idx / FOUT) % FIN;
  int hd = idx / (FOUT * FIN);
  wT[((size_t)hd * FOUT + o) * FIN + k] = f2bf(w[idx]);
}

// ---------------- phase 1: h_prime^T = (h @ w[h])^T, bf16 ----------------
// grid: (NN/16, NHEAD), block 256 (8 waves; wave w owns N-tile w)
__global__ void __launch_bounds__(256) gat_proj(
    const unsigned short* __restrict__ hb,   // [NN][FIN] bf16
    const unsigned short* __restrict__ wT,   // [H][FOUT][FIN] bf16
    unsigned short* __restrict__ hpT)        // [H][FOUT][NN] bf16
{
  const int lane = threadIdx.x & 31;
  const int wid  = threadIdx.x >> 5;
  const int m0   = blockIdx.x * 16;
  const int head = blockIdx.y;
  const int n0   = wid * 16;

  const int row   = m0 + (lane & 15);
  const int abase = (lane >> 4) * 8;               // A: K sub-base {0,8}
  const int col   = n0 + (lane & 15);
  const int bkoff = (lane >> 4) * 8;               // B: K sub-base {0,8}

  const unsigned short* arow = hb + (size_t)row * FIN;
  const unsigned short* bcol = wT + ((size_t)head * FOUT + col) * FIN;

  v8f c = {};
  for (int k0 = 0; k0 < FIN; k0 += 32) {
    FragU A, B;
    A.q[0] = *(const uint4*)(arow + k0 + abase);        // K = abase..abase+7
    A.q[1] = *(const uint4*)(arow + k0 + abase + 16);   // K = 16+abase..23+abase
    B.q[0] = *(const uint4*)(bcol + k0 + bkoff);        // K = bkoff..bkoff+7
    B.q[1] = *(const uint4*)(bcol + k0 + bkoff + 16);   // K = 16+bkoff..23+bkoff
    c = __builtin_amdgcn_wmma_f32_16x16x32_bf16(false, A.v, false, B.v,
                                                (short)0, c, false, false);
  }
  // C tile: VGPR r, lanes 0-15 -> M=r, lanes 16-31 -> M=r+8 ; col = n0+(lane&15)
  const int roff = (lane >> 4) * 8;
  union { unsigned short us[8]; uint4 q; } pk;
#pragma unroll
  for (int r = 0; r < 8; ++r) pk.us[r] = f2bf(c[r]);
  *(uint4*)(hpT + ((size_t)head * FOUT + col) * NN + m0 + roff) = pk.q;
}

// ---------------- attn_src / attn_dst ----------------
// grid: (NN/256, NHEAD), block 256
__global__ void gat_attn_vec(
    const unsigned short* __restrict__ hpT,
    const float* __restrict__ a_src, const float* __restrict__ a_dst,
    float* __restrict__ srcv, float* __restrict__ dstv)
{
  const int head = blockIdx.y;
  const int i = blockIdx.x * 256 + threadIdx.x;
  const unsigned short* base = hpT + (size_t)head * FOUT * NN + i;
  float as = 0.f, ad = 0.f;
  for (int o = 0; o < FOUT; ++o) {
    float v = bf2f(base[(size_t)o * NN]);
    as += v * a_src[head * FOUT + o];
    ad += v * a_dst[head * FOUT + o];
  }
  srcv[head * NN + i] = as;
  dstv[head * NN + i] = ad;
}

// ---------------- phase 2: fused masked-softmax + attn @ h_prime ----------------
// grid: NN/16 blocks, 256 threads (wave w = head w); 16 rows per block.
__global__ void __launch_bounds__(256) gat_attention(
    const int* __restrict__ adj,
    const unsigned short* __restrict__ hpT,
    const float* __restrict__ srcv, const float* __restrict__ dstv,
    const float* __restrict__ bias,
    float* __restrict__ outp,            // [NN][NHEAD*FOUT]
    float* __restrict__ weightp)         // [NN][NN]
{
  __shared__ __align__(16) float dlds[NHEAD][NN];        // 128 KB: all heads' dst
  __shared__ unsigned int abits[16][NN / 32];            // 8 KB: adj bitmask, 16 rows
  __shared__ float wchunk[16][33];                       // head-summed attn chunk

  const int lane = threadIdx.x & 31;
  const int head = threadIdx.x >> 5;                     // wave == head
  const int i0   = blockIdx.x * 16;

  // ---- pass 0a: stage dst vectors for all heads into LDS (coalesced) ----
  for (int idx = threadIdx.x; idx < (NHEAD * NN) / 4; idx += 256) {
    ((float4*)&dlds[0][0])[idx] = ((const float4*)dstv)[idx];
  }
  // ---- pass 0b: compress this block's 16 adj rows into an LDS bitmask ----
#if __has_builtin(__builtin_amdgcn_ballot_w32)
  for (int q = 0; q < 256; ++q) {                        // 8 waves x 256 dwords
    const int d   = head * 256 + q;                      // dword id 0..2047
    const int row = d >> 7;
    const int col = d & 127;
    const int a = adj[(size_t)(i0 + row) * NN + col * 32 + lane];
    const unsigned int m = __builtin_amdgcn_ballot_w32(a > 0);
    if (lane == 0) abits[row][col] = m;
  }
#else
  for (int idx = threadIdx.x; idx < 16 * (NN / 32); idx += 256) abits[idx >> 7][idx & 127] = 0u;
  __syncthreads();
  for (int e = threadIdx.x; e < 16 * NN; e += 256) {
    const int row = e >> 12, j = e & (NN - 1);
    if (adj[(size_t)(i0 + row) * NN + j] > 0) atomicOr(&abits[row][j >> 5], 1u << (j & 31));
  }
#endif
  __syncthreads();

  // ---- pass 1: per-row 1/sumexp for this head (no max pass: |s| small) ----
  float my_m = 0.f, my_rcl = 0.f;
  for (int r = 0; r < 16; ++r) {
    const float si = srcv[head * NN + i0 + r];
    float l = 0.f;
    for (int t = 0; t < NN / 128; ++t) {
      const int j = t * 128 + lane * 4;
      const unsigned int bits = abits[r][t * 4 + (lane >> 3)] >> ((lane & 7) * 4);
      float4 d = *(const float4*)&dlds[head][j];
      float s0 = si + d.x; s0 = s0 >= 0.f ? s0 : LEAKY * s0; s0 = (bits & 1u) ? s0 : NEGINF;
      float s1 = si + d.y; s1 = s1 >= 0.f ? s1 : LEAKY * s1; s1 = (bits & 2u) ? s1 : NEGINF;
      float s2 = si + d.z; s2 = s2 >= 0.f ? s2 : LEAKY * s2; s2 = (bits & 4u) ? s2 : NEGINF;
      float s3 = si + d.w; s3 = s3 >= 0.f ? s3 : LEAKY * s3; s3 = (bits & 8u) ? s3 : NEGINF;
      l += __expf(s0) + __expf(s1) + __expf(s2) + __expf(s3);
    }
    for (int off = 16; off; off >>= 1) l += __shfl_xor(l, off, 32);
    // fully-masked row: softmax of constant NEG_INF row is uniform 1/NN
    const float rcl = (l > 0.f) ? 1.0f / l : 1.0f / (float)NN;
    const float mo  = (l > 0.f) ? 0.f : NEGINF;
    if ((lane & 15) == r) { my_m = mo; my_rcl = rcl; }
  }

  // ---- pass 2: stream j in chunks of 32, WMMA-accumulate, head-sum weight ----
  const int r_a   = lane & 15;                 // A-fragment row owned by this lane
  const int kbase = (lane >> 4) * 8;           // A K sub-base {0,8}
  const int bkoff = (lane >> 4) * 8;           // B K sub-base {0,8}
  const unsigned short* hpTh = hpT + (size_t)head * FOUT * NN;
  const float s_i = srcv[head * NN + i0 + r_a];

  v8f cacc[8] = {};

  for (int jc = 0; jc < NN / 32; ++jc) {
    const int j0 = jc * 32;

    __syncthreads();
    {
      int idx = threadIdx.x;
      wchunk[idx >> 5][idx & 31] = 0.f;
      wchunk[(idx >> 5) + 8][idx & 31] = 0.f;
    }
    __syncthreads();

    const unsigned int bits = abits[r_a][jc];
    FragU A;
#pragma unroll
    for (int g = 0; g < 2; ++g) {
      const int jb = g * 16 + kbase;                 // 0..31 within chunk
      float4 d0 = *(const float4*)&dlds[head][j0 + jb];
      float4 d1 = *(const float4*)&dlds[head][j0 + jb + 4];
      const float dv[8] = {d0.x, d0.y, d0.z, d0.w, d1.x, d1.y, d1.z, d1.w};
#pragma unroll
      for (int u = 0; u < 8; ++u) {
        float s = s_i + dv[u];
        s = s >= 0.f ? s : LEAKY * s;
        s = ((bits >> (jb + u)) & 1u) ? s : NEGINF;
        float p = __expf(s - my_m) * my_rcl;
        atomicAdd(&wchunk[r_a][jb + u], p);          // sum over heads
        A.us[g * 8 + u] = f2bf(p);
      }
    }

#pragma unroll
    for (int t = 0; t < 8; ++t) {
      FragU B;
      const unsigned short* bp = hpTh + (size_t)(t * 16 + (lane & 15)) * NN + j0 + bkoff;
      B.q[0] = *(const uint4*)(bp);
      B.q[1] = *(const uint4*)(bp + 16);
      cacc[t] = __builtin_amdgcn_wmma_f32_16x16x32_bf16(false, A.v, false, B.v,
                                                        (short)0, cacc[t], false, false);
    }

    __syncthreads();
    {
      int idx = threadIdx.x;
      weightp[(size_t)(i0 + (idx >> 5)) * NN + j0 + (idx & 31)] = wchunk[idx >> 5][idx & 31];
      weightp[(size_t)(i0 + (idx >> 5) + 8) * NN + j0 + (idx & 31)] = wchunk[(idx >> 5) + 8][idx & 31];
    }
  }

  // ---- epilogue: out[i][head*128 + o] = cacc + bias ----
  const int roff = (lane >> 4) * 8;
#pragma unroll
  for (int t = 0; t < 8; ++t) {
    const int oc = t * 16 + (lane & 15);
    const float bv = bias[oc];
#pragma unroll
    for (int r = 0; r < 8; ++r) {
      outp[(size_t)(i0 + roff + r) * (NHEAD * FOUT) + head * FOUT + oc] = cacc[t][r] + bv;
    }
  }
}

// ---------------- host launcher ----------------
extern "C" void kernel_launch(void* const* d_in, const int* in_sizes, int n_in,
                              void* d_out, int out_size, void* d_ws, size_t ws_size,
                              hipStream_t stream) {
  const float* h     = (const float*)d_in[0];   // [4096,1024]
  const int*   adj   = (const int*)d_in[1];     // [4096,4096]
  const float* w     = (const float*)d_in[2];   // [8,1024,128]
  const float* bias  = (const float*)d_in[3];   // [128]
  const float* a_src = (const float*)d_in[4];   // [8,128,1]
  const float* a_dst = (const float*)d_in[5];   // [8,128,1]

  float* outp    = (float*)d_out;               // [4096, 1024]
  float* weightp = outp + (size_t)NN * (NHEAD * FOUT);  // [4096, 4096]

  char* ws = (char*)d_ws;
  const size_t HB_BYTES  = (size_t)NN * FIN * 2;          // 8 MB
  const size_t WT_BYTES  = (size_t)NHEAD * FOUT * FIN * 2;// 2 MB
  const size_t HPT_BYTES = (size_t)NHEAD * FOUT * NN * 2; // 8 MB
  const size_t SV_BYTES  = (size_t)NHEAD * NN * 4;        // 128 KB

  unsigned short* hb   = (unsigned short*)(ws);
  unsigned short* wT   = (unsigned short*)(ws + HB_BYTES);
  unsigned short* hpT  = (unsigned short*)(ws + HB_BYTES + WT_BYTES);
  float*          srcv = (float*)(ws + HB_BYTES + WT_BYTES + HPT_BYTES);
  float*          dstv = (float*)(ws + HB_BYTES + WT_BYTES + HPT_BYTES + SV_BYTES);

  gat_prep_h<<<(NN * FIN) / 256, 256, 0, stream>>>(h, hb);
  gat_prep_w<<<(NHEAD * FIN * FOUT) / 256, 256, 0, stream>>>(w, wT);
  gat_proj<<<dim3(NN / 16, NHEAD), 256, 0, stream>>>(hb, wT, hpT);
  gat_attn_vec<<<dim3(NN / 256, NHEAD), 256, 0, stream>>>(hpT, a_src, a_dst, srcv, dstv);
  gat_attention<<<NN / 16, 256, 0, stream>>>(adj, hpT, srcv, dstv, bias, outp, weightp);
}
